// Repro_11879879543878
// MI455X (gfx1250) — matrix operand order repaired and checked by reading, not compiled, not run on hardware
//
#include <hip/hip_runtime.h>
#include <hip/hip_bf16.h>

typedef _Float16 h16;
typedef __attribute__((ext_vector_type(16))) _Float16 v16h;
typedef __attribute__((ext_vector_type(4)))  _Float16 v4h;
typedef __attribute__((ext_vector_type(8)))  float    v8f;

#define NB   32
#define HH   64
#define CC   60
#define NHD  6
#define HD   10
#define WSZ  7
#define NWD  10      /* windows per spatial dim (70/7) */
#define NTOK 49
#define LOG100F 4.605170185988092f

#define QKSZ  (64 * 64 + 16)   /* q / k region size incl. over-read pad (4112 halves) */
#define VTOFF (2 * QKSZ)       /* v^T region offset (8224 halves)                     */

/* ---------------- fragment loaders (all loads unconditional) ----------------
 * A frag (16-bit 16x32): lane L -> M=m0+(L&15), K = k0 + 16*(L>>4) + i (contiguous)
 * B frag (16-bit 32x16): lane L -> N=n0+(L&15), K = k0 + 16*(L>>4) + i (contiguous)
 */
__device__ __forceinline__ v16h fragA(const h16* base, int ld, int m0, int k0, int lane) {
  const h16* p = base + (m0 + (lane & 15)) * ld + k0 + ((lane >> 4) << 4);
  v16h r;
#pragma unroll
  for (int i = 0; i < 8; ++i) r[i] = p[i];
#pragma unroll
  for (int i = 0; i < 8; ++i) r[8 + i] = p[8 + i];
  return r;
}

__device__ __forceinline__ v16h fragB(const h16* base, int ld, int n0, int k0, int lane) {
  const h16* p = base + (n0 + (lane & 15)) * ld + k0 + ((lane >> 4) << 4);
  v16h r;
#pragma unroll
  for (int i = 0; i < 16; ++i) r[i] = p[i];
  return r;
}

/* A frag over one head's 10-wide q block (base already offset by h*HD).
 * K>=16 statically zero; K 10..15 masked branchlessly after the load. */
__device__ __forceinline__ v16h fragA_h(const h16* base, int ld, int m0, int lane) {
  int kb = (lane >> 4) << 4;
  const h16* p = base + (m0 + (lane & 15)) * ld;
  v16h r;
#pragma unroll
  for (int i = 0; i < 16; ++i) {
    h16 t = p[i];                       /* unconditional load */
    r[i] = (kb + i < HD) ? t : (h16)0.0f;
  }
  return r;
}

/* B frag over one head's 10-wide k block (base already offset by h*HD). */
__device__ __forceinline__ v16h fragB_h(const h16* base, int ld, int n0, int lane) {
  int kb = (lane >> 4) << 4;
  const h16* p = base + (n0 + (lane & 15)) * ld;
  v16h r;
#pragma unroll
  for (int i = 0; i < 16; ++i) {
    h16 t = p[i];                       /* unconditional load */
    r[i] = (kb + i < HD) ? t : (h16)0.0f;
  }
  return r;
}

/* B frag from V^T rows (base = sQKV + VTOFF + h*HD*ld); N (=d) clamped per lane. */
__device__ __forceinline__ v16h fragB_vt(const h16* base, int ld, int k0, int lane) {
  int n = lane & 15;
  int nc = (n < HD) ? n : (HD - 1);
  const h16* p = base + nc * ld + k0 + ((lane >> 4) << 4);
  v16h r;
#pragma unroll
  for (int i = 0; i < 16; ++i) {
    h16 t = p[i];                       /* unconditional load */
    r[i] = (n < HD) ? t : (h16)0.0f;
  }
  return r;
}

/* score tiles for head h: S = (Q_h K_h^T) * iq_m * ik_n * exp(min(ls,log100)) */
__device__ __forceinline__ void s_tiles(const h16* sQKV, float* sS,
                                        const float* sIQ, const float* sIK,
                                        const float* lscale, int h, int lane,
                                        int t0, int tstep) {
  float sc = __expf(fminf(lscale[h], LOG100F));
  for (int t = t0; t < 16; t += tstep) {
    int mt = t >> 2, nt = t & 3;
    v16h a  = fragA_h(sQKV + h * HD, 64, mt * 16, lane);
    v16h bb = fragB_h(sQKV + QKSZ + h * HD, 64, nt * 16, lane);
    v8f acc = {};
    acc = __builtin_amdgcn_wmma_f32_16x16x32_f16(false, a, false, bb,
                                                 (short)0, acc, false, false);
    int n  = nt * 16 + (lane & 15);
    int mb = mt * 16 + ((lane >> 4) << 3);
    float kf = sIK[h * 64 + n] * sc;
#pragma unroll
    for (int j = 0; j < 8; ++j)
      sS[(mb + j) * 64 + n] = acc[j] * sIQ[h * 64 + mb + j] * kf;
  }
}

__global__ __launch_bounds__(256) void swinv2_win_attn(
    const float* __restrict__ x, const float* __restrict__ qkv_w,
    const float* __restrict__ qkv_b, const float* __restrict__ lscale,
    const float* __restrict__ proj_w, const float* __restrict__ proj_b,
    float* __restrict__ out) {
  __shared__ __align__(16) h16 sX[64 * 64];          /* x window -> later attn output  */
  /* q (stride 64) | k (stride 64) | v^T (stride 1): base offset is s*QKSZ            */
  __shared__ __align__(16) h16 sQKV[2 * QKSZ + 64 * 64];
  /* union: qkv weights (24576B) | scores f32 (16384B) + probs f16 (8192B);
   * proj weights f16 (8192B) reuse the (dead) score region for the final GEMM        */
  __shared__ __align__(16) unsigned char uni[24576];
  __shared__ float sIQ[NHD * 64];                    /* 1/|q| per (head, token), 0 pad */
  __shared__ float sIK[NHD * 64];
  h16*   sW  = (h16*)uni;                            /* [192][64] */
  float* sS  = (float*)uni;                          /* [64][64]  */
  h16*   sP  = (h16*)(uni + 16384);                  /* [64][64]  */
  h16*   sPW = (h16*)uni;                            /* [64][64]  */

  const int tid  = threadIdx.x;
  const int lane = tid & 31;
  const int wv   = tid >> 5;

  const int win  = blockIdx.x;                       /* 0..3199 */
  const int b    = win / (NWD * NWD);
  const int wrem = win % (NWD * NWD);
  const int wi   = wrem / NWD;
  const int wj   = wrem % NWD;

  /* warm L2 for the late-phase proj weights (global_prefetch_b8) */
  if (tid < 113) __builtin_prefetch(proj_w + tid * 32, 0, 1);

  /* ---- phase 1: vectorized staging (b128 loads, b64 stores), pads zeroed separately ---- */
  {
    const float4 zero4f = make_float4(0.f, 0.f, 0.f, 0.f);
    const float4* xv = (const float4*)x;             /* 49 tokens x 15 float4 */
    for (int p = tid; p < NTOK * 15; p += 256) {
      int t = p / 15, v4 = p % 15;
      int hr = wi * WSZ + t / WSZ;
      int wc = wj * WSZ + t % WSZ;
      bool ok = (hr < HH) && (wc < HH);
      int hrc = ok ? hr : 0;
      int wcc = ok ? wc : 0;
      float4 val = xv[((b * HH + hrc) * HH + wcc) * 15 + v4];   /* unconditional */
      v4h hv;
      hv[0] = ok ? (h16)val.x : (h16)0.0f;
      hv[1] = ok ? (h16)val.y : (h16)0.0f;
      hv[2] = ok ? (h16)val.z : (h16)0.0f;
      hv[3] = ok ? (h16)val.w : (h16)0.0f;
      *(v4h*)&sX[t * 64 + v4 * 4] = hv;
    }
    if (tid < 120)  ((float4*)(sX + NTOK * 64))[tid] = zero4f;   /* rows 49..63 */
    if (tid < NTOK) *(v4h*)&sX[tid * 64 + CC] = (v4h)(h16)0.0f;  /* cols 60..63 */

    const float4* wvv = (const float4*)qkv_w;        /* 180 rows x 15 float4 */
    for (int p = tid; p < 3 * CC * 15; p += 256) {
      int n = p / 15, v4 = p % 15;
      float4 val = wvv[p];
      v4h hv;
      hv[0] = (h16)val.x; hv[1] = (h16)val.y; hv[2] = (h16)val.z; hv[3] = (h16)val.w;
      *(v4h*)&sW[n * 64 + v4 * 4] = hv;
    }
    if (tid < 96)     ((float4*)(sW + 3 * CC * 64))[tid] = zero4f;
    if (tid < 3 * CC) *(v4h*)&sW[tid * 64 + CC] = (v4h)(h16)0.0f;
  }
  __syncthreads();

  /* ---- phase 2: qkv = X @ Wqkv^T + b, routed into q/k/v^T regions of sQKV ---- */
  for (int t = wv; t < 48; t += 8) {                 /* 4 M-tiles x 12 N-tiles */
    int mt = t & 3, nt = t >> 2;
    v8f acc = {};
#pragma unroll
    for (int ks = 0; ks < 2; ++ks) {
      v16h a  = fragA(sX, 64, mt * 16, ks * 32, lane);
      v16h bb = fragB(sW, 64, nt * 16, ks * 32, lane);
      acc = __builtin_amdgcn_wmma_f32_16x16x32_f16(false, a, false, bb,
                                                   (short)0, acc, false, false);
    }
    int n  = nt * 16 + (lane & 15);
    int mb = mt * 16 + ((lane >> 4) << 3);
    bool valid = n < 3 * CC;
    int nn = valid ? n : 0;
    float bias = qkv_b[nn];
    int s   = nn / CC;
    int rr  = nn - s * CC;                           /* rr = h*10+d in 0..59 */
    bool isv = (s == 2);                             /* v stored transposed  */
    int off  = s * QKSZ + (isv ? (rr << 6) : rr) + mb * (isv ? 1 : 64);
    int mstr = isv ? 1 : 64;
    if (valid) {
#pragma unroll
      for (int j = 0; j < 8; ++j) { sQKV[off] = (h16)(acc[j] + bias); off += mstr; }
    }
  }
  __syncthreads();

  /* ---- phase 3: inverse norms (pad entries = 0) + zero probs buffer ---- */
  for (int p = tid; p < NHD * 64; p += 256) {
    int h = p >> 6, t = p & 63;
    int tt = (t < NTOK) ? t : 0;
    const h16* q = &sQKV[tt * 64 + h * HD];
    const h16* k = &sQKV[QKSZ + tt * 64 + h * HD];
    float sq = 0.f, sk = 0.f;
#pragma unroll
    for (int d = 0; d < HD; ++d) {
      float a = (float)q[d], c = (float)k[d];
      sq += a * a; sk += c * c;
    }
    bool ok = t < NTOK;
    sIQ[p] = ok ? 1.f / fmaxf(sqrtf(sq), 1e-12f) : 0.f;
    sIK[p] = ok ? 1.f / fmaxf(sqrtf(sk), 1e-12f) : 0.f;
  }
  for (int i = tid; i < 64 * 16; i += 256) ((float4*)sP)[i] = make_float4(0.f, 0.f, 0.f, 0.f);
  __syncthreads();

  /* ---- phase 4: software-pipelined per-head attention ----
   * prologue: S(0) on all 8 waves; per head: softmax(h), then AV(h) on waves 0..3
   * overlapped with S(h+1) on waves 4..7 (or proj-weight staging at the last head). */
  s_tiles(sQKV, sS, sIQ, sIK, lscale, 0, lane, wv, 8);

  for (int h = 0; h < NHD; ++h) {
    __syncthreads();                                 /* sS(h) complete */
    if (tid < NTOK) {
      float mx = -1e30f;
      for (int j = 0; j < NTOK; ++j) mx = fmaxf(mx, sS[tid * 64 + j]);
      float sum = 0.f;
      for (int j = 0; j < NTOK; ++j) sum += __expf(sS[tid * 64 + j] - mx);
      float inv = 1.f / sum;
      for (int j = 0; j < NTOK; ++j)
        sP[tid * 64 + j] = (h16)(__expf(sS[tid * 64 + j] - mx) * inv);
    }
    __syncthreads();                                 /* sP(h) ready, sS free */

    if (wv < 4) {                                    /* O_h = P @ V_h */
      int mt = wv;
      v8f acc = {};
#pragma unroll
      for (int ks = 0; ks < 2; ++ks) {
        v16h a  = fragA(sP, 64, mt * 16, ks * 32, lane);
        v16h bb = fragB_vt(sQKV + VTOFF + h * HD * 64, 64, ks * 32, lane);
        acc = __builtin_amdgcn_wmma_f32_16x16x32_f16(false, a, false, bb,
                                                     (short)0, acc, false, false);
      }
      int d  = lane & 15;
      int mb = mt * 16 + ((lane >> 4) << 3);
      if (d < HD) {
#pragma unroll
        for (int j = 0; j < 8; ++j)
          sX[(mb + j) * 64 + h * HD + d] = (h16)acc[j];
      }
    } else if (h + 1 < NHD) {
      s_tiles(sQKV, sS, sIQ, sIK, lscale, h + 1, lane, wv - 4, 4);
    } else {
      /* stage proj weights into dead score region, zero-padded [64][64] */
      int lt = tid - 128;                            /* 0..127 */
      const float4* pwv = (const float4*)proj_w;
      for (int p = lt; p < CC * 15; p += 128) {
        int n = p / 15, v4 = p % 15;
        float4 val = pwv[p];
        v4h hv;
        hv[0] = (h16)val.x; hv[1] = (h16)val.y; hv[2] = (h16)val.z; hv[3] = (h16)val.w;
        *(v4h*)&sPW[n * 64 + v4 * 4] = hv;
      }
      if (lt < 32) ((float4*)(sPW + CC * 64))[lt] = make_float4(0.f, 0.f, 0.f, 0.f);
      if (lt < CC) *(v4h*)&sPW[lt * 64 + CC] = (v4h)(h16)0.0f;
    }
  }
  __syncthreads();

  /* ---- phase 5: proj + bias + window-reverse/unpad writeback ---- */
  for (int t = wv; t < 16; t += 8) {
    int mt = t >> 2, nt = t & 3;
    v8f acc = {};
#pragma unroll
    for (int ks = 0; ks < 2; ++ks) {
      v16h a  = fragA(sX, 64, mt * 16, ks * 32, lane);
      v16h bb = fragB(sPW, 64, nt * 16, ks * 32, lane);
      acc = __builtin_amdgcn_wmma_f32_16x16x32_f16(false, a, false, bb,
                                                   (short)0, acc, false, false);
    }
    int n  = nt * 16 + (lane & 15);
    int mb = mt * 16 + ((lane >> 4) << 3);
    if (n < CC) {
      float bias = proj_b[n];
#pragma unroll
      for (int j = 0; j < 8; ++j) {
        int m = mb + j;
        if (m < NTOK) {
          int hr = wi * WSZ + m / WSZ;
          int wc = wj * WSZ + m % WSZ;
          if (hr < HH && wc < HH)
            out[((b * HH + hr) * HH + wc) * CC + n] = acc[j] + bias;
        }
      }
    }
  }
}

extern "C" void kernel_launch(void* const* d_in, const int* in_sizes, int n_in,
                              void* d_out, int out_size, void* d_ws, size_t ws_size,
                              hipStream_t stream) {
  const float* x      = (const float*)d_in[0];
  const float* qkv_w  = (const float*)d_in[1];
  const float* qkv_b  = (const float*)d_in[2];
  const float* ls     = (const float*)d_in[3];
  const float* proj_w = (const float*)d_in[4];
  const float* proj_b = (const float*)d_in[5];
  float* out = (float*)d_out;

  dim3 grid(NB * NWD * NWD);   /* 3200 windows */
  dim3 block(256);             /* 8 wave32 */
  hipLaunchKernelGGL(swinv2_win_attn, grid, block, 0, stream,
                     x, qkv_w, qkv_b, ls, proj_w, proj_b, out);
}
